// NeuronBloomBlock_78726750535982
// MI455X (gfx1250) — matrix operand order repaired and checked by reading, not compiled, not run on hardware
//
#include <hip/hip_runtime.h>
#include <hip/hip_bf16.h>
#include <math.h>

// ---------------------------------------------------------------------------
// Types for CDNA5 WMMA (wave32): V_WMMA_F32_16X16X32_BF16
// ---------------------------------------------------------------------------
typedef __bf16        v8bf  __attribute__((ext_vector_type(8)));
typedef __bf16        v16bf __attribute__((ext_vector_type(16)));
typedef float         v8f   __attribute__((ext_vector_type(8)));
typedef unsigned int  v4u   __attribute__((ext_vector_type(4)));
typedef unsigned int  v8u   __attribute__((ext_vector_type(8)));

static __device__ __forceinline__ v16bf cat16(v8bf lo, v8bf hi) {
    return __builtin_shufflevector(lo, hi, 0,1,2,3,4,5,6,7,8,9,10,11,12,13,14,15);
}
static __device__ __forceinline__ v8f vzero8() {
    v8f z;
#pragma unroll
    for (int i = 0; i < 8; ++i) z[i] = 0.0f;
    return z;
}
static __device__ __forceinline__ v8f wmma_bf16(v16bf a, v16bf b, v8f c) {
    // 8 args: (neg_a, A, neg_b, B, c_mod, C, reuse_a, reuse_b)
    return __builtin_amdgcn_wmma_f32_16x16x32_bf16(false, a, false, b, (short)0, c, false, false);
}

// ---------------------------------------------------------------------------
// Tensor Data Mover: 2D tile load (global -> LDS) via D# descriptor.
//   CDNA5 ISA (group0) / §8.4 (group1).
//   2-group form (VADDR2/3 = NULL) for tensors up to 2D.
//   Padding done by TDM: pad_interval/pad_amount produce the bank-conflict-free
//   LDS row stride the WMMA fragment readers expect.
// ---------------------------------------------------------------------------
static __device__ __forceinline__ void tdm_load_2d(
    unsigned lds_addr,            // byte offset in LDS
    const void* gptr,             // global byte address of tile start
    unsigned tensor_d0,           // elements (fastest dim)
    unsigned tensor_d1,           // elements (rows)
    unsigned tile_d0,             // elements per row in tile
    unsigned tile_d1,             // rows in tile
    unsigned long long d0_stride, // row stride in elements
    unsigned data_size_code,      // 0:1B 1:2B 2:4B 3:8B
    unsigned pad_en,              // 1 = pad LDS rows
    unsigned pad_interval_code,   // 0=2 DW,1=4,2=8,3=16,4=32,...
    unsigned pad_amount_code)     // (DWORDs of pad) - 1
{
    const unsigned long long ga = (unsigned long long)gptr;
    v4u g0;
    g0[0] = 1u;                                             // count=1 (valid D#)
    g0[1] = lds_addr;                                       // lds_addr
    g0[2] = (unsigned)(ga & 0xffffffffu);                   // global_addr[31:0]
    g0[3] = (unsigned)((ga >> 32) & 0x01ffffffu) | (2u << 30); // [56:32] | type=2
    v8u g1;
    g1[0] = (data_size_code << 16) | (pad_en << 20) |
            (pad_interval_code << 22) | (pad_amount_code << 25);
    g1[1] = (tensor_d0 & 0xffffu) << 16;                    // tensor_dim0[15:0]
    g1[2] = (tensor_d0 >> 16) | ((tensor_d1 & 0xffffu) << 16);
    g1[3] = (tensor_d1 >> 16) | (tile_d0 << 16);            // tile_dim0
    g1[4] = tile_d1;                                        // tile_dim1 (tile_dim2=0)
    g1[5] = (unsigned)(d0_stride & 0xffffffffu);            // dim0_stride[31:0]
    g1[6] = (unsigned)((d0_stride >> 32) & 0xffffu);        // dim0_stride[47:32]
    g1[7] = 0u;                                             // dim1_stride (unused, 2D)
    asm volatile("tensor_load_to_lds %0, %1" :: "s"(g0), "s"(g1) : "memory");
}

// ---------------------------------------------------------------------------
// fp32 -> bf16 transposed weight conversion: out[n*K + k] = (bf16)in[k*N + n]
// ---------------------------------------------------------------------------
__global__ __launch_bounds__(256) void cvt_t_kernel(const float* __restrict__ in,
                                                    __bf16* __restrict__ out,
                                                    int K, int N) {
    long long idx = (long long)blockIdx.x * 256 + threadIdx.x;
    if (idx >= (long long)K * N) return;
    int k = (int)(idx / N);
    int n = (int)(idx % N);
    out[(size_t)n * K + k] = (__bf16)in[idx];
}

// ---------------------------------------------------------------------------
// LayerNorm over D (one block per row), bf16 output for WMMA GEMM input
// ---------------------------------------------------------------------------
__global__ __launch_bounds__(256) void ln_bf16_kernel(const float* __restrict__ x,
                                                      const float* __restrict__ w,
                                                      const float* __restrict__ b,
                                                      __bf16* __restrict__ out,
                                                      int Dsz) {
    __shared__ float red[256];
    const int tid = threadIdx.x;
    const float* xr = x + (size_t)blockIdx.x * Dsz;

    float s = 0.0f;
    for (int i = tid; i < Dsz; i += 256) s += xr[i];
    red[tid] = s; __syncthreads();
    for (int off = 128; off > 0; off >>= 1) {
        if (tid < off) red[tid] += red[tid + off];
        __syncthreads();
    }
    const float mu = red[0] / (float)Dsz;
    __syncthreads();

    float v = 0.0f;
    for (int i = tid; i < Dsz; i += 256) { float d = xr[i] - mu; v += d * d; }
    red[tid] = v; __syncthreads();
    for (int off = 128; off > 0; off >>= 1) {
        if (tid < off) red[tid] += red[tid + off];
        __syncthreads();
    }
    const float rstd = rsqrtf(red[0] / (float)Dsz + 1e-5f);

    __bf16* orow = out + (size_t)blockIdx.x * Dsz;
    for (int i = tid; i < Dsz; i += 256)
        orow[i] = (__bf16)((xr[i] - mu) * rstd * w[i] + b[i]);
}

// ---------------------------------------------------------------------------
// WMMA GEMM with TDM double-buffered staging:
//   C[M,N] = act(A[M,K] @ Bt[N,K]^T + bias) (+ residual)
//   A : bf16 row-major [M][K],  Bt : bf16 row-major [N][K] (W transposed)
//   128x128 workgroup tile, 8 waves (2x4), each wave 64x32 (4x2 WMMA tiles)
//   BK = 64 (two bf16 WMMA k-steps per TDM stage)
//   TDM pads each 128-byte row with 16 bytes -> 72-element LDS row stride
//   (bank stride 36 mod 64 => 16 distinct banks for fragment reads).
// ---------------------------------------------------------------------------
#define GBM 128
#define GBN 128
#define GBK 64
#define LPAD 8   // elements of TDM pad per row (4 DWORDs)

__global__ __launch_bounds__(256) void gemm_bf16_kernel(
    const __bf16* __restrict__ A,
    const __bf16* __restrict__ Bt,
    const float*  __restrict__ bias,
    const float*  __restrict__ res,   // may be null
    float*        __restrict__ Cf,    // may be null
    __bf16*       __restrict__ Cb,    // may be null
    int M, int N, int K, int act)     // act: 0 = none, 1 = tanh-GELU
{
    __shared__ __attribute__((aligned(16))) __bf16 As[2][GBM][GBK + LPAD];
    __shared__ __attribute__((aligned(16))) __bf16 Bs[2][GBN][GBK + LPAD];

    const int tid  = threadIdx.x;
    const int lane = tid & 31;
    const int wave = tid >> 5;
    const int wm   = wave & 1;    // wave row    (0..1) -> 64 rows
    const int wn   = wave >> 1;   // wave column (0..3) -> 32 cols
    const int half = lane >> 4;   // 0/1 (lane group)
    const int l16  = lane & 15;

    const int m0 = blockIdx.y * GBM;
    const int n0 = blockIdx.x * GBN;

    v8f acc[4][2];
#pragma unroll
    for (int mi = 0; mi < 4; ++mi)
#pragma unroll
        for (int ni = 0; ni < 2; ++ni) acc[mi][ni] = vzero8();

    const unsigned ldsA0 = (unsigned)(uintptr_t)&As[0][0][0];
    const unsigned ldsA1 = (unsigned)(uintptr_t)&As[1][0][0];
    const unsigned ldsB0 = (unsigned)(uintptr_t)&Bs[0][0][0];
    const unsigned ldsB1 = (unsigned)(uintptr_t)&Bs[1][0][0];

    const int ntiles = K / GBK;

    // prologue: stage tile 0 into buffer 0 (wave 0 drives the TDM)
    if (wave == 0) {
        tdm_load_2d(ldsA0, A + (size_t)m0 * K, GBK, GBM, GBK, GBM,
                    (unsigned long long)K, 1u, 1u, 4u, 3u);
        tdm_load_2d(ldsB0, Bt + (size_t)n0 * K, GBK, GBN, GBK, GBN,
                    (unsigned long long)K, 1u, 1u, 4u, 3u);
    }

    int buf = 0;
    for (int t = 0; t < ntiles; ++t, buf ^= 1) {
        if (wave == 0) __builtin_amdgcn_s_wait_tensorcnt(0);
        __syncthreads();                       // buffer `buf` is now readable

        if (wave == 0 && (t + 1) < ntiles) {   // stream next stage into buf^1
            const size_t koff = (size_t)(t + 1) * GBK;
            tdm_load_2d(buf ? ldsA0 : ldsA1, A + (size_t)m0 * K + koff,
                        GBK, GBM, GBK, GBM, (unsigned long long)K, 1u, 1u, 4u, 3u);
            tdm_load_2d(buf ? ldsB0 : ldsB1, Bt + (size_t)n0 * K + koff,
                        GBK, GBN, GBK, GBN, (unsigned long long)K, 1u, 1u, 4u, 3u);
        }

        // compute: 2 k-steps of 32, 16 WMMAs per wave per stage
#pragma unroll
        for (int ks = 0; ks < 2; ++ks) {
            v16bf bfr[2];
#pragma unroll
            for (int ni = 0; ni < 2; ++ni) {
                const __bf16* p = &Bs[buf][wn * 32 + ni * 16 + l16][ks * 32 + half * 8];
                bfr[ni] = cat16(*(const v8bf*)p, *(const v8bf*)(p + 16));
            }
#pragma unroll
            for (int mi = 0; mi < 4; ++mi) {
                const __bf16* p = &As[buf][wm * 64 + mi * 16 + l16][ks * 32 + half * 8];
                v16bf af = cat16(*(const v8bf*)p, *(const v8bf*)(p + 16));
                acc[mi][0] = wmma_bf16(af, bfr[0], acc[mi][0]);
                acc[mi][1] = wmma_bf16(af, bfr[1], acc[mi][1]);
            }
        }
        // no trailing barrier needed: next iteration's barrier (after the
        // tensorcnt drain) separates this compute from the next overwrite.
    }

    // epilogue — C layout: row = tile + half*8 + r, col = tile + l16
#pragma unroll
    for (int mi = 0; mi < 4; ++mi) {
#pragma unroll
        for (int ni = 0; ni < 2; ++ni) {
            const int colg = n0 + wn * 32 + ni * 16 + l16;
            const float bv = bias[colg];
#pragma unroll
            for (int r = 0; r < 8; ++r) {
                const int rowg = m0 + wm * 64 + mi * 16 + half * 8 + r;
                float v = acc[mi][ni][r] + bv;
                if (act == 1) {
                    const float u = 0.7978845608028654f * (v + 0.044715f * v * v * v);
                    v = 0.5f * v * (1.0f + tanhf(u));
                }
                const size_t o = (size_t)rowg * N + colg;
                if (res) v += res[o];
                if (Cf)  Cf[o] = v;
                if (Cb)  Cb[o] = (__bf16)v;
            }
        }
    }
}

// ---------------------------------------------------------------------------
// Flash attention with ALiBi + causal mask, all-WMMA (QK^T and P·V).
//   qkv : bf16 [B][S][3*D], head h: q at col h*128, k at D+h*128, v at 2D+h*128
//   out : bf16 [B][S][D]
//   Grid: (S/64, B*H), 128 threads (4 waves, 16 query rows per wave)
// ---------------------------------------------------------------------------
__global__ __launch_bounds__(128) void attn_kernel(
    const __bf16* __restrict__ qkv,
    __bf16* __restrict__ out,
    int Bsz, int Ssz, int Dsz, int Hh)
{
    const int HD = 128;
    __shared__ __attribute__((aligned(16))) __bf16 Ks[32][136];     // [key][dim]
    __shared__ __attribute__((aligned(16))) __bf16 Vt[128][40];     // [dim][key]
    __shared__ __attribute__((aligned(16))) __bf16 Pld[4][16][40];  // per-wave P

    const int tid  = threadIdx.x;
    const int lane = tid & 31;
    const int wave = tid >> 5;
    const int half = lane >> 4;
    const int l16  = lane & 15;

    const int b = blockIdx.y / Hh;
    const int h = blockIdx.y % Hh;
    const int q0    = blockIdx.x * 64;
    const int qrow0 = q0 + wave * 16;

    const size_t rstride = (size_t)3 * Dsz;
    const __bf16* qbase = qkv + (size_t)b * Ssz * rstride + (size_t)h * HD;
    const __bf16* kbase = qbase + Dsz;
    const __bf16* vbase = qbase + 2 * (size_t)Dsz;

    const float scale = 0.08838834764831845f;           // 1/sqrt(128)
    const float slope = exp2f(-0.5f * (float)(h + 1));  // ALiBi, H = 16

    // Q fragments (A-layout), held in registers for the whole row-block
    v16bf qf[4];
#pragma unroll
    for (int ks = 0; ks < 4; ++ks) {
        const __bf16* p = qbase + (size_t)(qrow0 + l16) * rstride + ks * 32 + half * 8;
        qf[ks] = cat16(*(const v8bf*)p, *(const v8bf*)(p + 16));
    }

    v8f o[8];
#pragma unroll
    for (int od = 0; od < 8; ++od) o[od] = vzero8();
    float mrow[8], lrow[8];
#pragma unroll
    for (int r = 0; r < 8; ++r) { mrow[r] = -1e30f; lrow[r] = 0.0f; }

    const int jend = q0 + 64;                 // causal: keys in [0, q0+63]
    for (int j0 = 0; j0 < jend; j0 += 32) {
        // ---- cooperative K/V staging (32 keys x 128 dims) ----
        {
            const int key = tid >> 2;         // 0..31
            const int cs  = (tid & 3) * 32;   // 0,32,64,96
            const __bf16* kg = kbase + (size_t)(j0 + key) * rstride + cs;
            const __bf16* vg = vbase + (size_t)(j0 + key) * rstride + cs;
#pragma unroll
            for (int u = 0; u < 4; ++u) {
                *(v8bf*)&Ks[key][cs + u * 8] = *(const v8bf*)(kg + u * 8);
                v8bf vv = *(const v8bf*)(vg + u * 8);
#pragma unroll
                for (int e = 0; e < 8; ++e) Vt[cs + u * 8 + e][key] = vv[e];
            }
        }
        __syncthreads();

        // ---- scores: S = Q K^T (two 16-key tiles) ----
        v8f s[2];
        s[0] = vzero8(); s[1] = vzero8();
#pragma unroll
        for (int ni = 0; ni < 2; ++ni) {
#pragma unroll
            for (int ks = 0; ks < 4; ++ks) {
                const __bf16* p = &Ks[ni * 16 + l16][ks * 32 + half * 8];
                v16bf kf = cat16(*(const v8bf*)p, *(const v8bf*)(p + 16));
                s[ni] = wmma_bf16(qf[ks], kf, s[ni]);
            }
        }

        // ---- online softmax (ALiBi + causal), P staged to LDS as bf16 ----
#pragma unroll
        for (int r = 0; r < 8; ++r) {
            const int irow = qrow0 + half * 8 + r;
            const int jc0 = j0 + l16;
            const int jc1 = jc0 + 16;
            float s0 = s[0][r] * scale + slope * (float)(jc0 - irow);
            float s1 = s[1][r] * scale + slope * (float)(jc1 - irow);
            if (jc0 > irow) s0 = -1e30f;
            if (jc1 > irow) s1 = -1e30f;

            float mx = fmaxf(s0, s1);
#pragma unroll
            for (int off = 1; off < 16; off <<= 1)
                mx = fmaxf(mx, __shfl_xor(mx, off, 32));
            const float nm = fmaxf(mrow[r], mx);
            const float p0 = expf(s0 - nm);
            const float p1 = expf(s1 - nm);
            float ps = p0 + p1;
#pragma unroll
            for (int off = 1; off < 16; off <<= 1)
                ps += __shfl_xor(ps, off, 32);
            const float alpha = expf(mrow[r] - nm);
            lrow[r] = lrow[r] * alpha + ps;
            mrow[r] = nm;

            Pld[wave][half * 8 + r][l16]      = (__bf16)p0;
            Pld[wave][half * 8 + r][16 + l16] = (__bf16)p1;
#pragma unroll
            for (int od = 0; od < 8; ++od) o[od][r] *= alpha;
        }

        // ---- P (C-layout -> A-layout via LDS), then O += P @ V ----
        v16bf pf;
        {
            const __bf16* p = &Pld[wave][l16][half * 8];
            pf = cat16(*(const v8bf*)p, *(const v8bf*)(p + 16));
        }
#pragma unroll
        for (int od = 0; od < 8; ++od) {
            const __bf16* p = &Vt[od * 16 + l16][half * 8];
            v16bf vf = cat16(*(const v8bf*)p, *(const v8bf*)(p + 16));
            o[od] = wmma_bf16(pf, vf, o[od]);
        }
        __syncthreads();
    }

    // ---- normalize and store (bf16, [B][S][D] at head column h*128) ----
#pragma unroll
    for (int od = 0; od < 8; ++od) {
#pragma unroll
        for (int r = 0; r < 8; ++r) {
            const int irow = qrow0 + half * 8 + r;
            const int col  = od * 16 + l16;
            out[((size_t)b * Ssz + irow) * Dsz + h * HD + col] =
                (__bf16)(o[od][r] / lrow[r]);
        }
    }
}

// ---------------------------------------------------------------------------
// Host-side orchestration
// ---------------------------------------------------------------------------
extern "C" void kernel_launch(void* const* d_in, const int* in_sizes, int n_in,
                              void* d_out, int out_size, void* d_ws, size_t ws_size,
                              hipStream_t stream) {
    (void)in_sizes; (void)n_in; (void)out_size; (void)ws_size;
    const int Bc = 2, Sc = 2048, Dc = 2048, Hc = 16, FFc = 8192;
    const int Mr = Bc * Sc;  // 4096 rows

    const float* x     = (const float*)d_in[0];
    const float* ln1_w = (const float*)d_in[1];
    const float* ln1_b = (const float*)d_in[2];
    const float* wqkv  = (const float*)d_in[3];
    const float* bqkv  = (const float*)d_in[4];
    const float* wo    = (const float*)d_in[5];
    const float* bo    = (const float*)d_in[6];
    const float* ln2_w = (const float*)d_in[7];
    const float* ln2_b = (const float*)d_in[8];
    const float* w1    = (const float*)d_in[9];
    const float* b1    = (const float*)d_in[10];
    const float* w2    = (const float*)d_in[11];
    const float* b2    = (const float*)d_in[12];
    float* outp = (float*)d_out;

    char* ws = (char*)d_ws;
    size_t off = 0;
    auto take = [&](size_t bytes) -> void* {
        void* p = ws + off;
        off = (off + bytes + 255) & ~(size_t)255;
        return p;
    };
    __bf16* wqkv_t = (__bf16*)take((size_t)Dc * 3 * Dc * 2);
    __bf16* wo_t   = (__bf16*)take((size_t)Dc * Dc * 2);
    __bf16* w1_t   = (__bf16*)take((size_t)Dc * FFc * 2);
    __bf16* w2_t   = (__bf16*)take((size_t)FFc * Dc * 2);
    __bf16* h1     = (__bf16*)take((size_t)Mr * Dc * 2);
    __bf16* qkvb   = (__bf16*)take((size_t)Mr * 3 * Dc * 2);
    __bf16* attnb  = (__bf16*)take((size_t)Mr * Dc * 2);
    float*  x1f    = (float*) take((size_t)Mr * Dc * 4);
    __bf16* h2     = (__bf16*)take((size_t)Mr * Dc * 2);
    __bf16* m1     = (__bf16*)take((size_t)Mr * FFc * 2);

    // 1) weights -> bf16, transposed to [N][K]
    {
        long long n;
        n = (long long)Dc * 3 * Dc;
        cvt_t_kernel<<<(unsigned)((n + 255) / 256), 256, 0, stream>>>(wqkv, wqkv_t, Dc, 3 * Dc);
        n = (long long)Dc * Dc;
        cvt_t_kernel<<<(unsigned)((n + 255) / 256), 256, 0, stream>>>(wo, wo_t, Dc, Dc);
        n = (long long)Dc * FFc;
        cvt_t_kernel<<<(unsigned)((n + 255) / 256), 256, 0, stream>>>(w1, w1_t, Dc, FFc);
        n = (long long)FFc * Dc;
        cvt_t_kernel<<<(unsigned)((n + 255) / 256), 256, 0, stream>>>(w2, w2_t, FFc, Dc);
    }

    // 2) LN1 -> h1 (bf16)
    ln_bf16_kernel<<<Mr, 256, 0, stream>>>(x, ln1_w, ln1_b, h1, Dc);

    // 3) QKV GEMM -> qkvb (bf16)
    gemm_bf16_kernel<<<dim3(3 * Dc / GBN, Mr / GBM), 256, 0, stream>>>(
        h1, wqkv_t, bqkv, nullptr, nullptr, qkvb, Mr, 3 * Dc, Dc, 0);

    // 4) flash attention -> attnb (bf16)
    attn_kernel<<<dim3(Sc / 64, Bc * Hc), 128, 0, stream>>>(qkvb, attnb, Bc, Sc, Dc, Hc);

    // 5) WO GEMM + residual(x) -> x1f (fp32)
    gemm_bf16_kernel<<<dim3(Dc / GBN, Mr / GBM), 256, 0, stream>>>(
        attnb, wo_t, bo, x, x1f, nullptr, Mr, Dc, Dc, 0);

    // 6) LN2 -> h2 (bf16)
    ln_bf16_kernel<<<Mr, 256, 0, stream>>>(x1f, ln2_w, ln2_b, h2, Dc);

    // 7) MLP up + GELU -> m1 (bf16)
    gemm_bf16_kernel<<<dim3(FFc / GBN, Mr / GBM), 256, 0, stream>>>(
        h2, w1_t, b1, nullptr, nullptr, m1, Mr, FFc, Dc, 1);

    // 8) MLP down + residual(x1f) -> d_out (fp32)
    gemm_bf16_kernel<<<dim3(Dc / GBN, Mr / GBM), 256, 0, stream>>>(
        m1, w2_t, b2, x1f, outp, nullptr, Mr, Dc, FFc, 0);
}
